// UnifiedHybridAttention_68015102099512
// MI455X (gfx1250) — matrix operand order repaired and checked by reading, not compiled, not run on hardware
//
#include <hip/hip_runtime.h>
#include <hip/hip_bf16.h>

#define T_    1024
#define D_    512
#define H_    8
#define C_    64
#define DQ_   256
#define P_    128
#define LMAX_ 16
#define NWIN_ 128
#define TOPK_ 32
#define NI_   2
#define CI_   64
#define NEG_  (-1e30f)
#define ROPE_BASE_ 10000.0f

typedef __attribute__((ext_vector_type(16))) __bf16       bf16x16;
typedef __attribute__((ext_vector_type(8)))  float        f32x8;
typedef __attribute__((ext_vector_type(4)))  unsigned int u32x4;
typedef int v4i __attribute__((vector_size(16)));   // pointee type for async-LDS builtin

#if defined(__HIP_DEVICE_COMPILE__) && __has_builtin(__builtin_amdgcn_global_load_async_to_lds_b128)
#define HAVE_ASYNC_LDS 1
#else
#define HAVE_ASYNC_LDS 0
#endif

struct Frag { union { bf16x16 v; u32x4 u[2]; }; };

__device__ __forceinline__ void load_a_frag(Frag& f, const __hip_bfloat16* arow, int k0)
{
    f.u[0] = *reinterpret_cast<const u32x4*>(arow + k0);        // K k0..+7   (+half*8)
    f.u[1] = *reinterpret_cast<const u32x4*>(arow + k0 + 16);   // K k0+16..+23
}
__device__ __forceinline__ void load_b_frag(Frag& f, const __hip_bfloat16* brow, int k0)
{
    f.u[0] = *reinterpret_cast<const u32x4*>(brow + k0);        // K k0..+7   (+half*8)
    f.u[1] = *reinterpret_cast<const u32x4*>(brow + k0 + 16);   // K k0+16..+23
}

// ---------------------------------------------------------------------------
// Register-blocked, double-buffered bf16 WMMA GEMM:
//   C[M,N](f32) = A[M,K](bf16, row major) x Bt[N,K]^T.
// Block = 4 waves; each wave owns one 16-row tile and 4 accumulators
// covering a 16x64 output strip (A fragment reused 4x per K-step).
// Loads for K-step k+32 are issued before the 4 WMMAs of step k so VMEM
// latency overlaps matrix issue.
// A-operand layout (16-bit, 16x32): lane l<16 -> row l, K {k0..+7, k0+16..+23};
// lane l+16 -> row l, K {+8..+15, +24..+31}.
// B-operand layout: lane n<16 -> col n, K {k0..+7, k0+16..+23}; lane n+16 -> K {+8..+15, +24..+31}.
// C/D: VGPR j -> M=j (lanes 0-15) / M=j+8 (lanes 16-31), lane -> N.
// ---------------------------------------------------------------------------
__global__ __launch_bounds__(128) void gemm_bf16_wmma(
    const __hip_bfloat16* __restrict__ A,
    const __hip_bfloat16* __restrict__ Bt,
    float* __restrict__ C,
    int M, int N, int K)
{
    const int wave = threadIdx.y;                 // 0..3 -> M sub-tile
    const int tm   = blockIdx.x * 4 + wave;
    const int tn0  = blockIdx.y * 4;              // 4 consecutive N tiles
    const int lane = threadIdx.x;
    const int half = lane >> 4;                   // 0 or 1
    const int l15  = lane & 15;

    const __hip_bfloat16* arow  = A  + (size_t)(tm * 16 + l15) * K + half * 8;
    const __hip_bfloat16* bbase = Bt + (size_t)(tn0 * 16 + l15) * K + half * 8;
    const size_t bstride = (size_t)16 * K;        // elements between N tiles

    f32x8 acc[4] = {};

    Frag fa, fb[4];
    load_a_frag(fa, arow, 0);
#pragma unroll
    for (int jt = 0; jt < 4; ++jt) load_b_frag(fb[jt], bbase + jt * bstride, 0);

    int k0 = 0;
    for (; k0 + 32 < K; k0 += 32) {
        Frag na, nb[4];
        const int kn = k0 + 32;
        load_a_frag(na, arow, kn);                 // issue next-step loads first
#pragma unroll
        for (int jt = 0; jt < 4; ++jt) load_b_frag(nb[jt], bbase + jt * bstride, kn);
        __builtin_prefetch(arow + kn + 32, 0, 0);
        __builtin_prefetch(bbase + kn + 32, 0, 0);
#pragma unroll
        for (int jt = 0; jt < 4; ++jt)             // compute current step
            acc[jt] = __builtin_amdgcn_wmma_f32_16x16x32_bf16(
                          false, fa.v, false, fb[jt].v, (short)0, acc[jt], false, false);
        fa = na;
#pragma unroll
        for (int jt = 0; jt < 4; ++jt) fb[jt] = nb[jt];
    }
#pragma unroll
    for (int jt = 0; jt < 4; ++jt)                 // last K-step
        acc[jt] = __builtin_amdgcn_wmma_f32_16x16x32_bf16(
                      false, fa.v, false, fb[jt].v, (short)0, acc[jt], false, false);

    const int cm = tm * 16 + half * 8;
#pragma unroll
    for (int jt = 0; jt < 4; ++jt) {
        const int cn = (tn0 + jt) * 16 + l15;
#pragma unroll
        for (int j = 0; j < 8; ++j)
            C[(size_t)(cm + j) * N + cn] = acc[jt][j];
    }
}

// --------------------------- f32 -> bf16 (row major) -----------------------
__global__ void cvt_bf16(const float* __restrict__ x, __hip_bfloat16* __restrict__ y, int n)
{
    int i = blockIdx.x * blockDim.x + threadIdx.x;
    if (i < n) y[i] = __float2bfloat16(x[i]);
}

// ------------------ f32 W[K,N] -> bf16 Wt[N,K] (transpose) -----------------
__global__ void tcvt_bf16(const float* __restrict__ W, __hip_bfloat16* __restrict__ Wt,
                          int K, int N)
{
    int i = blockIdx.x * blockDim.x + threadIdx.x;
    if (i >= K * N) return;
    int k = i / N, n = i % N;
    Wt[(size_t)n * K + k] = __float2bfloat16(W[(size_t)k * N + n]);
}

// --------------------------- phrase row gather -----------------------------
__global__ void gather_rows(const __hip_bfloat16* __restrict__ hb,
                            const int* __restrict__ idx,
                            __hip_bfloat16* __restrict__ hp,
                            int rows, int Dd)
{
    int i = blockIdx.x * blockDim.x + threadIdx.x;
    if (i >= rows * Dd) return;
    int r = i / Dd, d = i % Dd;
    hp[i] = hb[(size_t)idx[r] * Dd + d];
}

// -------- masked softmax over LMAX and weighted combine (per channel) ------
// z,cv: [P, LMAX, F], bias: [LMAX, F], mask: [P, LMAX] (u8) -> out [P, F]
__global__ void combine_softmax(const float* __restrict__ z,
                                const float* __restrict__ cv,
                                const float* __restrict__ bias,
                                const unsigned char* __restrict__ mask,
                                float* __restrict__ out,
                                int Pn, int F)
{
    int i = blockIdx.x * blockDim.x + threadIdx.x;
    if (i >= Pn * F) return;
    int p = i / F, f = i % F;

    float zv[LMAX_];
    float m = NEG_;
    bool any = false;
#pragma unroll
    for (int l = 0; l < LMAX_; ++l) {
        bool ok = mask[p * LMAX_ + l] != 0;
        any |= ok;
        float v = ok ? (z[((size_t)p * LMAX_ + l) * F + f] + bias[l * F + f]) : NEG_;
        zv[l] = v;
        m = fmaxf(m, v);
    }
    float s = 0.f;
#pragma unroll
    for (int l = 0; l < LMAX_; ++l) s += __expf(zv[l] - m);
    float o = 0.f;
#pragma unroll
    for (int l = 0; l < LMAX_; ++l)
        o += (__expf(zv[l] - m) / s) * cv[((size_t)p * LMAX_ + l) * F + f];
    out[i] = any ? o : 0.f;
}

// ------------------ fused RMSNorm (optional) + RoPE on 64-ch rows ----------
// x,out: [rows, 64].  pos = posArr ? clamp(posArr[r/div]) : r/div.
__global__ void rmsrope(const float* __restrict__ x, float* __restrict__ out,
                        const float* __restrict__ nw,
                        const int* __restrict__ posArr, int div,
                        int rows, int Tmax)
{
    int r = blockIdx.x * blockDim.x + threadIdx.x;
    if (r >= rows) return;
    const float* xi = x + (size_t)r * 64;

    float buf[64];
    float ss = 0.f;
#pragma unroll
    for (int c = 0; c < 64; ++c) { float v = xi[c]; buf[c] = v; ss += v * v; }
    if (nw) {
        float inv = rsqrtf(ss * (1.f / 64.f) + 1e-6f);
#pragma unroll
        for (int c = 0; c < 64; ++c) buf[c] = buf[c] * inv * nw[c];
    }
    int pos = posArr ? posArr[r / div] : (r / div);
    pos = min(max(pos, 0), Tmax - 1);
    float fp = (float)pos;
#pragma unroll
    for (int c = 0; c < 64; ++c) {
        int   cc = c & 31;
        float fr = fp * __powf(ROPE_BASE_, -(float)cc * (1.f / 32.f));
        float cs = __cosf(fr), sn = __sinf(fr);
        float rot = (c < 32) ? -buf[c + 32] : buf[c - 32];
        out[(size_t)r * 64 + c] = buf[c] * cs + rot * sn;
    }
}

// ---------------------------- h @ W_w (tiny) -------------------------------
__global__ void hww(const float* __restrict__ h, const float* __restrict__ Ww,
                    float* __restrict__ w)
{
    int i = blockIdx.x * blockDim.x + threadIdx.x;
    if (i >= T_ * NI_) return;
    int t = i / NI_, ni = i % NI_;
    float s = 0.f;
    for (int d = 0; d < D_; ++d) s += h[(size_t)t * D_ + d] * Ww[d * NI_ + ni];
    w[i] = s;
}

// ------------- indexer scores I[t,p] = sum_i relu(q_i . k_idx)*w ----------
__global__ void score(const float* __restrict__ q_i,    // [T, NI*CI]
                      const float* __restrict__ k_idx,  // [P, CI]
                      const float* __restrict__ w,      // [T, NI]
                      const int*   __restrict__ endp,   // [P]
                      float* __restrict__ I)            // [T, P]
{
    int i = blockIdx.x * blockDim.x + threadIdx.x;
    if (i >= T_ * P_) return;
    int t = i / P_, p = i % P_;
    float s = 0.f;
#pragma unroll
    for (int ni = 0; ni < NI_; ++ni) {
        float d = 0.f;
        for (int c = 0; c < CI_; ++c)
            d += q_i[(size_t)t * (NI_ * CI_) + ni * CI_ + c] * k_idx[p * CI_ + c];
        s += fmaxf(d, 0.f) * w[t * NI_ + ni];
    }
    int e = endp[p];
    I[i] = (e <= t && e >= 0) ? s : NEG_;
}

// ----------------------- top-32 of 128 per token ---------------------------
__global__ __launch_bounds__(128) void topk(const float* __restrict__ I,
                                            int* __restrict__ sel,
                                            int* __restrict__ selok)
{
    __shared__ float sv[P_];
    __shared__ float rv[P_];
    __shared__ int   ri[P_];
    int t = blockIdx.x, j = threadIdx.x;
    sv[j] = I[(size_t)t * P_ + j];
    __syncthreads();
    for (int k = 0; k < TOPK_; ++k) {
        rv[j] = sv[j]; ri[j] = j;
        __syncthreads();
        for (int off = 64; off > 0; off >>= 1) {
            if (j < off && rv[j + off] > rv[j]) { rv[j] = rv[j + off]; ri[j] = ri[j + off]; }
            __syncthreads();
        }
        if (j == 0) {
            sel[t * TOPK_ + k]   = ri[0];
            selok[t * TOPK_ + k] = (rv[0] > NEG_ * 0.5f) ? 1 : 0;
            sv[ri[0]] = -3e38f;
        }
        __syncthreads();
    }
}

// --------- fused attention: 32 selected + 128 window + sink, softmax -------
// Window K rows are staged into LDS with CDNA5 async global->LDS loads.
__global__ __launch_bounds__(64) void attention(
    const float* __restrict__ q,      // [T, H*C]   (post rms+rope)
    const float* __restrict__ k_csa,  // [P, H*C]
    const float* __restrict__ c_csa,  // [P, H*C]
    const float* __restrict__ k_sw,   // [T, H*C]
    const float* __restrict__ kv,     // [T, H*C]   (v for window)
    const int*   __restrict__ sel,    // [T, 32]
    const int*   __restrict__ selok,  // [T, 32]
    const float* __restrict__ sink,   // [H]
    __hip_bfloat16* __restrict__ out_bf)  // [T, H*C]
{
    __shared__ float kwin[NWIN_ * C_];            // 32 KB window-K stage
    __shared__ float lg[TOPK_ + NWIN_ + 1];
    __shared__ float red[2];
    int t = blockIdx.x, hh = blockIdx.y, j = threadIdx.x;
    const float scale = 0.125f;                   // 1/sqrt(64)
    const float* qv = q + (size_t)t * (H_ * C_) + hh * C_;

    // ---- stage window K rows into LDS ----
    for (int wi = j; wi < NWIN_; wi += 64) {
        int src = t - (NWIN_ - 1) + wi;
        if (src >= 0) {
            const float* g = k_sw + ((size_t)src * H_ + hh) * C_;
            float*       l = kwin + wi * C_;
#if HAVE_ASYNC_LDS
#pragma unroll
            for (int c = 0; c < C_; c += 4)
                __builtin_amdgcn_global_load_async_to_lds_b128(
                    (__attribute__((address_space(1))) v4i*)(g + c),
                    (__attribute__((address_space(3))) v4i*)(l + c), 0, 0);
#else
#pragma unroll
            for (int c = 0; c < C_; ++c) l[c] = g[c];
#endif
        }
    }
#if HAVE_ASYNC_LDS
#if __has_builtin(__builtin_amdgcn_s_wait_asynccnt)
    __builtin_amdgcn_s_wait_asynccnt(0);
#else
    asm volatile("s_wait_asynccnt 0" ::: "memory");
#endif
#endif
    __syncthreads();

    // ---- logits ----
    for (int k = j; k < TOPK_ + NWIN_; k += 64) {
        float d = NEG_;
        if (k < TOPK_) {
            if (selok[t * TOPK_ + k]) {
                int p = sel[t * TOPK_ + k];
                const float* kk = k_csa + ((size_t)p * H_ + hh) * C_;
                float a = 0.f;
                for (int c = 0; c < C_; ++c) a += qv[c] * kk[c];
                d = a * scale;
            }
        } else {
            int wi = k - TOPK_;
            if (t - (NWIN_ - 1) + wi >= 0) {
                const float* kk = kwin + wi * C_;
                float a = 0.f;
                for (int c = 0; c < C_; ++c) a += qv[c] * kk[c];
                d = a * scale;
            }
        }
        lg[k] = d;
    }
    if (j == 0) lg[TOPK_ + NWIN_] = sink[hh];
    __syncthreads();
    if (j == 0) {
        float m = NEG_;
        for (int k = 0; k <= TOPK_ + NWIN_; ++k) m = fmaxf(m, lg[k]);
        float s = 0.f;
        for (int k = 0; k <= TOPK_ + NWIN_; ++k) s += __expf(lg[k] - m);
        red[0] = m; red[1] = s;
    }
    __syncthreads();
    float m = red[0], invs = 1.f / red[1];

    float o = 0.f;                                // output channel j (j<64)
    for (int k = 0; k < TOPK_; ++k) {
        float pb = __expf(lg[k] - m) * invs;
        int p = sel[t * TOPK_ + k];               // pb==0 when !selok
        o += pb * c_csa[((size_t)p * H_ + hh) * C_ + j];
    }
    for (int k = 0; k < NWIN_; ++k) {
        float pb = __expf(lg[TOPK_ + k] - m) * invs;
        int src = max(t - (NWIN_ - 1) + k, 0);    // pb==0 when src<0
        o += pb * kv[((size_t)src * H_ + hh) * C_ + j];
    }
    out_bf[(size_t)t * (H_ * C_) + hh * C_ + j] = __float2bfloat16(o);
}

// ===========================================================================
extern "C" void kernel_launch(void* const* d_in, const int* in_sizes, int n_in,
                              void* d_out, int out_size, void* d_ws, size_t ws_size,
                              hipStream_t stream)
{
    const float*         h     = (const float*)d_in[0];
    const unsigned char* mask  = (const unsigned char*)d_in[1];
    const int*           tok   = (const int*)d_in[2];
    const int*           endp  = (const int*)d_in[3];
    /* d_in[4] phrase_id unused */
    const float* W_dq   = (const float*)d_in[5];
    const float* W_uq   = (const float*)d_in[6];
    const float* W_ckv  = (const float*)d_in[7];
    const float* W_cz   = (const float*)d_in[8];
    const float* B_csa  = (const float*)d_in[9];
    const float* W_ikv  = (const float*)d_in[10];
    const float* W_iz   = (const float*)d_in[11];
    const float* B_idx  = (const float*)d_in[12];
    const float* W_iuq  = (const float*)d_in[13];
    const float* W_w    = (const float*)d_in[14];
    const float* W_swkv = (const float*)d_in[15];
    const float* qnw    = (const float*)d_in[16];
    const float* knw    = (const float*)d_in[17];
    const float* W_o    = (const float*)d_in[18];
    const float* sink   = (const float*)d_in[19];
    float* out = (float*)d_out;

    char* ws = (char*)d_ws;
    size_t off = 0;
    auto alloc = [&](size_t bytes) -> char* {
        char* p = ws + off;
        off = (off + bytes + 255) & ~(size_t)255;
        return p;
    };
    typedef __hip_bfloat16 bf;

    bf* h_bf     = (bf*)alloc((size_t)T_ * D_ * 2);
    bf* hp_bf    = (bf*)alloc((size_t)P_ * LMAX_ * D_ * 2);
    bf* Wt_dq    = (bf*)alloc((size_t)DQ_ * D_ * 2);
    bf* Wt_uq    = (bf*)alloc((size_t)(H_ * C_) * DQ_ * 2);
    bf* Wt_ckv   = (bf*)alloc((size_t)D_ * D_ * 2);
    bf* Wt_cz    = (bf*)alloc((size_t)D_ * D_ * 2);
    bf* Wt_ikv   = (bf*)alloc((size_t)CI_ * D_ * 2);
    bf* Wt_iz    = (bf*)alloc((size_t)CI_ * D_ * 2);
    bf* Wt_iuq   = (bf*)alloc((size_t)(NI_ * CI_) * DQ_ * 2);
    bf* Wt_swkv  = (bf*)alloc((size_t)D_ * D_ * 2);
    bf* Wt_o     = (bf*)alloc((size_t)D_ * D_ * 2);
    float* q_lat = (float*)alloc((size_t)T_ * DQ_ * 4);
    bf* q_lat_bf = (bf*)alloc((size_t)T_ * DQ_ * 2);
    float* qbuf  = (float*)alloc((size_t)T_ * H_ * C_ * 4);
    float* c_tok = (float*)alloc((size_t)P_ * LMAX_ * D_ * 4);
    float* zbuf  = (float*)alloc((size_t)P_ * LMAX_ * D_ * 4);
    float* ck    = (float*)alloc((size_t)P_ * LMAX_ * CI_ * 4);
    float* zk    = (float*)alloc((size_t)P_ * LMAX_ * CI_ * 4);
    float* c_csa = (float*)alloc((size_t)P_ * H_ * C_ * 4);
    float* k_csa = (float*)alloc((size_t)P_ * H_ * C_ * 4);
    float* ki_raw= (float*)alloc((size_t)P_ * CI_ * 4);
    float* k_idx = (float*)alloc((size_t)P_ * CI_ * 4);
    float* q_i   = (float*)alloc((size_t)T_ * NI_ * CI_ * 4);
    float* wbuf  = (float*)alloc((size_t)T_ * NI_ * 4);
    float* Ibuf  = (float*)alloc((size_t)T_ * P_ * 4);
    int*   sel   = (int*)alloc((size_t)T_ * TOPK_ * 4);
    int*   selok = (int*)alloc((size_t)T_ * TOPK_ * 4);
    float* kvbuf = (float*)alloc((size_t)T_ * H_ * C_ * 4);
    float* k_sw  = (float*)alloc((size_t)T_ * H_ * C_ * 4);
    bf* attn_bf  = (bf*)alloc((size_t)T_ * H_ * C_ * 2);

    auto cdiv = [](int a, int b) { return (a + b - 1) / b; };
    const dim3 gblk(32, 4);   // 4 waves per GEMM block (64x64 C tile)

    // ---- precision conversion + weight transposes ----
    cvt_bf16<<<cdiv(T_ * D_, 256), 256, 0, stream>>>(h, h_bf, T_ * D_);
    tcvt_bf16<<<cdiv(D_ * DQ_, 256), 256, 0, stream>>>(W_dq, Wt_dq, D_, DQ_);
    tcvt_bf16<<<cdiv(DQ_ * H_ * C_, 256), 256, 0, stream>>>(W_uq, Wt_uq, DQ_, H_ * C_);
    tcvt_bf16<<<cdiv(D_ * D_, 256), 256, 0, stream>>>(W_ckv, Wt_ckv, D_, D_);
    tcvt_bf16<<<cdiv(D_ * D_, 256), 256, 0, stream>>>(W_cz, Wt_cz, D_, D_);
    tcvt_bf16<<<cdiv(D_ * CI_, 256), 256, 0, stream>>>(W_ikv, Wt_ikv, D_, CI_);
    tcvt_bf16<<<cdiv(D_ * CI_, 256), 256, 0, stream>>>(W_iz, Wt_iz, D_, CI_);
    tcvt_bf16<<<cdiv(DQ_ * NI_ * CI_, 256), 256, 0, stream>>>(W_iuq, Wt_iuq, DQ_, NI_ * CI_);
    tcvt_bf16<<<cdiv(D_ * D_, 256), 256, 0, stream>>>(W_swkv, Wt_swkv, D_, D_);
    tcvt_bf16<<<cdiv(D_ * D_, 256), 256, 0, stream>>>(W_o, Wt_o, D_, D_);

    // ---- q path: q_lat = h@W_dq ; q = q_lat@W_uq ; rms+rope ----
    gemm_bf16_wmma<<<dim3(T_ / 64, DQ_ / 64), gblk, 0, stream>>>(h_bf, Wt_dq, q_lat, T_, DQ_, D_);
    cvt_bf16<<<cdiv(T_ * DQ_, 256), 256, 0, stream>>>(q_lat, q_lat_bf, T_ * DQ_);
    gemm_bf16_wmma<<<dim3(T_ / 64, (H_ * C_) / 64), gblk, 0, stream>>>(q_lat_bf, Wt_uq, qbuf, T_, H_ * C_, DQ_);
    rmsrope<<<cdiv(T_ * H_, 256), 256, 0, stream>>>(qbuf, qbuf, qnw, (const int*)nullptr, H_, T_ * H_, T_);

    // ---- phrase path: gather, c_tok/z/ck/zk GEMMs, combine, rms+rope ----
    gather_rows<<<cdiv(P_ * LMAX_ * D_, 256), 256, 0, stream>>>(h_bf, tok, hp_bf, P_ * LMAX_, D_);
    gemm_bf16_wmma<<<dim3((P_ * LMAX_) / 64, D_ / 64), gblk, 0, stream>>>(hp_bf, Wt_ckv, c_tok, P_ * LMAX_, D_, D_);
    gemm_bf16_wmma<<<dim3((P_ * LMAX_) / 64, D_ / 64), gblk, 0, stream>>>(hp_bf, Wt_cz, zbuf, P_ * LMAX_, D_, D_);
    gemm_bf16_wmma<<<dim3((P_ * LMAX_) / 64, CI_ / 64), gblk, 0, stream>>>(hp_bf, Wt_ikv, ck, P_ * LMAX_, CI_, D_);
    gemm_bf16_wmma<<<dim3((P_ * LMAX_) / 64, CI_ / 64), gblk, 0, stream>>>(hp_bf, Wt_iz, zk, P_ * LMAX_, CI_, D_);
    combine_softmax<<<cdiv(P_ * H_ * C_, 256), 256, 0, stream>>>(zbuf, c_tok, B_csa, mask, c_csa, P_, H_ * C_);
    combine_softmax<<<cdiv(P_ * CI_, 256), 256, 0, stream>>>(zk, ck, B_idx, mask, ki_raw, P_, CI_);
    rmsrope<<<cdiv(P_ * H_, 256), 256, 0, stream>>>(c_csa, k_csa, knw, endp, H_, P_ * H_, T_);
    rmsrope<<<cdiv(P_, 256), 256, 0, stream>>>(ki_raw, k_idx, (const float*)nullptr, endp, 1, P_, T_);

    // ---- indexer scores + top-k ----
    gemm_bf16_wmma<<<dim3(T_ / 64, (NI_ * CI_) / 64), gblk, 0, stream>>>(q_lat_bf, Wt_iuq, q_i, T_, NI_ * CI_, DQ_);
    hww<<<cdiv(T_ * NI_, 256), 256, 0, stream>>>(h, W_w, wbuf);
    score<<<cdiv(T_ * P_, 256), 256, 0, stream>>>(q_i, k_idx, wbuf, endp, Ibuf);
    topk<<<T_, 128, 0, stream>>>(Ibuf, sel, selok);

    // ---- sliding window K/V ----
    gemm_bf16_wmma<<<dim3(T_ / 64, D_ / 64), gblk, 0, stream>>>(h_bf, Wt_swkv, kvbuf, T_, D_, D_);
    rmsrope<<<cdiv(T_ * H_, 256), 256, 0, stream>>>(kvbuf, k_sw, knw, (const int*)nullptr, H_, T_ * H_, T_);

    // ---- fused attention (selected + window + sink) ----
    attention<<<dim3(T_, H_), 64, 0, stream>>>(qbuf, k_csa, c_csa, k_sw, kvbuf, sel, selok, sink, attn_bf);

    // ---- output projection ----
    gemm_bf16_wmma<<<dim3(T_ / 64, D_ / 64), gblk, 0, stream>>>(attn_bf, Wt_o, out, T_, D_, D_);

    (void)in_sizes; (void)n_in; (void)out_size; (void)ws_size;
}
